// GQA_26671746908805
// MI455X (gfx1250) — matrix operand order repaired and checked by reading, not compiled, not run on hardware
//
#include <hip/hip_runtime.h>

// ---------------------------------------------------------------------------
// GQA attention layer for MI455X (gfx1250, wave32, WMMA 16x16x32 bf16)
//
// Pipeline:
//   1) cast x -> bf16; pack wq|wk|wv transposed -> [3072][2048] bf16,
//      wo transposed -> [2048][2048] bf16
//   2) QKV GEMM: double-buffered GLOBAL_LOAD_ASYNC_TO_LDS_B128 staging,
//      bf16 WMMA with f32 accum
//   3) per-(token,head) RMSNorm + RoPE in f32 (attention scale folded into Q),
//      scatter to head-major bf16 (V stored transposed [d][s])
//   4) flash attention: 1 wave per 16-row Q tile, 32-key blocks,
//      online softmax in f32, QK^T and PV via v_wmma_f32_16x16x32_bf16
//   5) output GEMM (same async-staged WMMA kernel) -> f32 d_out
// ---------------------------------------------------------------------------

typedef __bf16 v16bf __attribute__((ext_vector_type(16)));
typedef __bf16 v8bf  __attribute__((ext_vector_type(8)));
typedef float  v8f   __attribute__((ext_vector_type(8)));

#define D_MODEL 2048
#define N_HEADS 16
#define N_KV    4
#define DH      128
#define SEQ     2048
#define BATCH   2
#define TOKENS  (BATCH * SEQ)     // 4096
#define QKV_N   3072
#define ATT_SCALE 0.08838834764831845f   // 1/sqrt(128)

__device__ __forceinline__ v8f wmma_bf16(v16bf a, v16bf b, v8f c) {
  // D = A(16x32 bf16) * B(32x16 bf16) + C(16x16 f32)
  return __builtin_amdgcn_wmma_f32_16x16x32_bf16(false, a, false, b,
                                                 (short)0, c, false, false);
}

__device__ __forceinline__ v16bf cat8(v8bf lo, v8bf hi) {
  return __builtin_shufflevector(lo, hi, 0, 1, 2, 3, 4, 5, 6, 7,
                                 8, 9, 10, 11, 12, 13, 14, 15);
}

// A-fragment (16x32, bf16) from row-major src with leading dim ld.
// Layout: lane<16 holds row=lane, K = 0..15; lane>=16 holds
// row=lane-16, K = 16..31.
__device__ __forceinline__ v16bf load_a(const __bf16* src, int ld, int row0,
                                        int k0, int lane) {
  const __bf16* p =
      src + (size_t)(row0 + (lane & 15)) * ld + k0 + ((lane >> 4) << 4);
  return *(const v16bf*)p;
}

// B-fragment (32x16, bf16) where column n of B is row n of bt ([N][K] layout).
// Layout (contiguous k, as in load_a): lanes 0-15 hold col=lane, K=0..15; lanes 16-31 hold col=lane-16,
// K=16..31 (contiguous per lane -> one 32-byte load).
__device__ __forceinline__ v16bf load_b(const __bf16* bt, int ld, int n0,
                                        int k0, int lane) {
  const __bf16* p =
      bt + (size_t)(n0 + (lane & 15)) * ld + k0 + ((lane >> 4) << 4);
  return *(const v16bf*)p;
}

// CDNA5 async global->LDS copy (ASYNCcnt-tracked, no VGPR round trip).
// Generic pointers to __shared__ carry the LDS byte offset in their low 32
// bits (ISA 10.2), which is what VDST wants.
__device__ __forceinline__ void async_load_b128(void* lds_ptr,
                                                const void* gptr) {
  unsigned lds_off = (unsigned)(uintptr_t)lds_ptr;
  asm volatile("global_load_async_to_lds_b128 %0, %1, off"
               :: "v"(lds_off), "v"(gptr)
               : "memory");
}
__device__ __forceinline__ void wait_async0() {
  asm volatile("s_wait_asynccnt 0x0" ::: "memory");
}

// ---------------------------------------------------------------------------
// Pre-pass casts / packing
// ---------------------------------------------------------------------------
__global__ void cvt_bf16_kernel(const float* __restrict__ src,
                                __bf16* __restrict__ dst, int n) {
  int i = blockIdx.x * blockDim.x + threadIdx.x;
  if (i < n) dst[i] = (__bf16)src[i];
}

// wqkvT[n][k] = concat(wq, wk, wv)[k][n]  (bf16, [3072][2048])
__global__ void pack_wqkvT_kernel(const float* __restrict__ wq,
                                  const float* __restrict__ wk,
                                  const float* __restrict__ wv,
                                  __bf16* __restrict__ dst) {
  int i = blockIdx.x * blockDim.x + threadIdx.x;   // < 3072*2048
  int n = i >> 11, k = i & 2047;
  float v;
  if (n < 2048)       v = wq[(size_t)k * 2048 + n];
  else if (n < 2560)  v = wk[(size_t)k * 512 + (n - 2048)];
  else                v = wv[(size_t)k * 512 + (n - 2560)];
  dst[i] = (__bf16)v;
}

// woT[n][k] = wo[k][n]  (bf16, [2048][2048])
__global__ void pack_woT_kernel(const float* __restrict__ wo,
                                __bf16* __restrict__ dst) {
  int i = blockIdx.x * blockDim.x + threadIdx.x;   // < 2048*2048
  int n = i >> 11, k = i & 2047;
  dst[i] = (__bf16)wo[(size_t)k * 2048 + n];
}

// ---------------------------------------------------------------------------
// bf16 WMMA GEMM: C[M,N](f32) = A[M,K](bf16) * Bt[N,K](bf16)^T
// Block: 256 threads = 8 waves; block tile 128x128; wave tile 32x64.
// A/B 128x32 tiles double-buffered in LDS via async global->LDS loads.
// ---------------------------------------------------------------------------
__global__ __launch_bounds__(256) void gemm_bf16_kernel(
    const __bf16* __restrict__ A, const __bf16* __restrict__ Bt,
    float* __restrict__ C, int N, int K) {
  __shared__ __align__(128) __bf16 Ab[2][128 * 32];
  __shared__ __align__(128) __bf16 Bb[2][128 * 32];
  int tid = threadIdx.x;
  int wave = tid >> 5, lane = tid & 31;
  int bm = blockIdx.y * 128, bn = blockIdx.x * 128;
  int wm = (wave >> 1) * 32;   // LDS-relative row of this wave's A tile
  int wn = (wave & 1) * 64;    // LDS-relative row of this wave's B tile

  // Stage one 128x32 A tile + one 128x32 B tile (8 KB each) into buffer buf.
  // 512 x 16-byte chunks per tile; 256 threads x 2 issues each.
  auto stage = [&](int buf, int k0) {
#pragma unroll
    for (int i = 0; i < 2; ++i) {
      int c = i * 256 + tid;           // 16B chunk id within tile
      int row = c >> 2;                // 4 chunks per 64B row
      int off = (c & 3) << 3;          // element offset within row
      async_load_b128(&Ab[buf][row * 32 + off],
                      A + (size_t)(bm + row) * K + k0 + off);
      async_load_b128(&Bb[buf][row * 32 + off],
                      Bt + (size_t)(bn + row) * K + k0 + off);
    }
  };

  stage(0, 0);
  wait_async0();
  __syncthreads();

  v8f acc[2][4] = {};
  int nk = K >> 5;
  for (int ki = 0; ki < nk; ++ki) {
    int buf = ki & 1;
    if (ki + 1 < nk) stage(buf ^ 1, (ki + 1) << 5);   // prefetch next tile

    const __bf16* Abuf = Ab[buf];
    const __bf16* Bbuf = Bb[buf];
    v16bf a0 = load_a(Abuf, 32, wm, 0, lane);
    v16bf a1 = load_a(Abuf, 32, wm + 16, 0, lane);
    v16bf b0 = load_b(Bbuf, 32, wn + 0, 0, lane);
    v16bf b1 = load_b(Bbuf, 32, wn + 16, 0, lane);
    v16bf b2 = load_b(Bbuf, 32, wn + 32, 0, lane);
    v16bf b3 = load_b(Bbuf, 32, wn + 48, 0, lane);
    acc[0][0] = wmma_bf16(a0, b0, acc[0][0]);
    acc[0][1] = wmma_bf16(a0, b1, acc[0][1]);
    acc[0][2] = wmma_bf16(a0, b2, acc[0][2]);
    acc[0][3] = wmma_bf16(a0, b3, acc[0][3]);
    acc[1][0] = wmma_bf16(a1, b0, acc[1][0]);
    acc[1][1] = wmma_bf16(a1, b1, acc[1][1]);
    acc[1][2] = wmma_bf16(a1, b2, acc[1][2]);
    acc[1][3] = wmma_bf16(a1, b3, acc[1][3]);

    wait_async0();     // this wave's prefetch landed in LDS
    __syncthreads();   // every wave's prefetch landed, reads of buf done
  }

  // C layout: element e of v8f -> row (e + 8*(lane>=16)), col (lane&15)
  int r = lane & 15, h8 = (lane >> 4) << 3;
#pragma unroll
  for (int i = 0; i < 2; ++i)
#pragma unroll
    for (int j = 0; j < 4; ++j)
#pragma unroll
      for (int e = 0; e < 8; ++e)
        C[(size_t)(bm + wm + 16 * i + h8 + e) * N + bn + wn + 16 * j + r] =
            acc[i][j][e];
}

// ---------------------------------------------------------------------------
// RMSNorm + RoPE + head scatter. One wave per (token, slot):
// slot 0..15 -> Q head, 16..19 -> K head, 20..23 -> V head.
// Lane owns 4 consecutive d-values (d = 4*lane .. 4*lane+3) of DH=128.
// Q is pre-multiplied by 1/sqrt(DH) (commutes with the RoPE rotation).
// ---------------------------------------------------------------------------
__global__ __launch_bounds__(256) void qkv_post_kernel(
    const float* __restrict__ qkv, const float* __restrict__ qw,
    const float* __restrict__ kw, __bf16* __restrict__ qh,
    __bf16* __restrict__ kh, __bf16* __restrict__ vT) {
  int wave = threadIdx.x >> 5, lane = threadIdx.x & 31;
  int u = blockIdx.x * 8 + wave;
  int token = u / 24, slot = u - token * 24;
  int b = token >> 11, s = token & 2047;
  int base = (slot < 16) ? slot * 128
                         : (slot < 20 ? 2048 + (slot - 16) * 128
                                      : 2560 + (slot - 20) * 128);
  const float* row = qkv + (size_t)token * 3072 + base;
  float v[4];
#pragma unroll
  for (int i = 0; i < 4; ++i) v[i] = row[4 * lane + i];

  if (slot >= 20) {  // V: plain cast, stored transposed [d][s]
    int kvh = slot - 20;
    __bf16* dst = vT + (size_t)((b * N_KV + kvh) * DH) * SEQ + s;
#pragma unroll
    for (int i = 0; i < 4; ++i)
      dst[(size_t)(4 * lane + i) * SEQ] = (__bf16)v[i];
    return;
  }

  // RMSNorm over DH=128 (sum of squares across the full wave32)
  float ss = v[0] * v[0] + v[1] * v[1] + v[2] * v[2] + v[3] * v[3];
#pragma unroll
  for (int off = 16; off >= 1; off >>= 1) ss += __shfl_xor(ss, off, 32);
  float rn = rsqrtf(ss * (1.0f / 128.0f) + 1e-6f);
  const float* w = (slot < 16) ? qw : kw;
#pragma unroll
  for (int i = 0; i < 4; ++i) v[i] *= rn * w[4 * lane + i];

  // RoPE: element d pairs with d^64; lanes<16 hold first half, >=16 second.
  float sp = (float)s;
  float sc = (slot < 16) ? ATT_SCALE : 1.0f;
  float o[4];
#pragma unroll
  for (int i = 0; i < 4; ++i) {
    int j = 4 * (lane & 15) + i;  // frequency index = d mod 64
    float ang = sp * __expf(-(float)j * 0.14391156510305212f); // ln(1e4)/64
    float c = __cosf(ang), sn = __sinf(ang);
    float p = __shfl_xor(v[i], 16, 32);   // partner half
    o[i] = sc * ((lane < 16) ? (v[i] * c - p * sn) : (v[i] * c + p * sn));
  }

  __bf16* dst = (slot < 16)
      ? qh + ((size_t)(b * N_HEADS + slot) * SEQ + s) * DH
      : kh + ((size_t)(b * N_KV + (slot - 16)) * SEQ + s) * DH;
#pragma unroll
  for (int i = 0; i < 4; ++i) dst[4 * lane + i] = (__bf16)o[i];
}

// ---------------------------------------------------------------------------
// Causal flash attention. One wave owns a 16-row Q tile; processes keys in
// 32-wide blocks: 8 WMMA for QK^T, online softmax in f32, P restaged through
// per-wave LDS into A-fragment layout (s_wait_dscnt for the RAW), 8 WMMA for
// P*V (V pre-transposed so B-fragments are contiguous). K/V are L2-resident
// (1 MB per (b,kv) group) so direct global fragment loads are sufficient.
// ---------------------------------------------------------------------------
__global__ __launch_bounds__(256) void attn_kernel(
    const __bf16* __restrict__ qh, const __bf16* __restrict__ kh,
    const __bf16* __restrict__ vT, __bf16* __restrict__ oh) {
  __shared__ __align__(32) __bf16 plds[8][16 * 32];
  int wave = threadIdx.x >> 5, lane = threadIdx.x & 31;
  int r = lane & 15, h8 = (lane >> 4) << 3;
  int h = blockIdx.y, b = blockIdx.z;
  int qbase = blockIdx.x * 128 + wave * 16;
  int kvh = h >> 2;  // N_GROUPS = 4
  const __bf16* Q  = qh + ((size_t)(b * N_HEADS + h) * SEQ + qbase) * DH;
  const __bf16* Kp = kh + (size_t)(b * N_KV + kvh) * SEQ * DH;
  const __bf16* Vt = vT + (size_t)(b * N_KV + kvh) * DH * SEQ;
  __bf16* pl = plds[wave];

  // Q tile 16x128 = 4 A-fragments, held in registers for the whole loop
  v16bf qf[4];
#pragma unroll
  for (int d = 0; d < 4; ++d) qf[d] = load_a(Q, DH, 0, d * 32, lane);

  float m[8], l[8];
  v8f acc[8] = {};
#pragma unroll
  for (int e = 0; e < 8; ++e) { m[e] = -1e30f; l[e] = 0.0f; }

  int nkb = (qbase + 47) >> 5;  // 32-key blocks covering keys <= qbase+15
  for (int kb = 0; kb < nkb; ++kb) {
    int sb = kb << 5;
    // --- scores: two 16x16 tiles over this 32-key block ---
    v8f sc0 = {}, sc1 = {};
#pragma unroll
    for (int d = 0; d < 4; ++d) {
      v16bf kf0 = load_b(Kp, DH, sb, d * 32, lane);
      v16bf kf1 = load_b(Kp, DH, sb + 16, d * 32, lane);
      sc0 = wmma_bf16(qf[d], kf0, sc0);
      sc1 = wmma_bf16(qf[d], kf1, sc1);
    }
    bool last = (kb == nkb - 1);
    // --- online softmax (row stats live per v8f element, reduced over the
    //     16 lanes of each half via shfl_xor 1/2/4/8) ---
#pragma unroll
    for (int e = 0; e < 8; ++e) {
      float s0 = sc0[e];
      float s1 = sc1[e];
      if (last) {
        int rowg = qbase + h8 + e;
        if (sb + r > rowg)      s0 = -1e30f;
        if (sb + 16 + r > rowg) s1 = -1e30f;
      }
      float mx = fmaxf(s0, s1);
#pragma unroll
      for (int off = 8; off >= 1; off >>= 1)
        mx = fmaxf(mx, __shfl_xor(mx, off, 32));
      float mn = fmaxf(m[e], mx);
      float alpha = __expf(m[e] - mn);
      float p0 = __expf(s0 - mn), p1 = __expf(s1 - mn);
      float rs = p0 + p1;
#pragma unroll
      for (int off = 8; off >= 1; off >>= 1) rs += __shfl_xor(rs, off, 32);
      l[e] = l[e] * alpha + rs;
      m[e] = mn;
#pragma unroll
      for (int d = 0; d < 8; ++d) acc[d][e] *= alpha;
      // stage P (bf16) row-major 16x32 in this wave's LDS slice
      pl[(h8 + e) * 32 + r]      = (__bf16)p0;
      pl[(h8 + e) * 32 + 16 + r] = (__bf16)p1;
    }
    // wave-local LDS RAW: waves diverge in trip count, so no block barrier --
    // use the CDNA5 split counter directly.
    asm volatile("s_wait_dscnt 0x0" ::: "memory");
    const __bf16* pp = pl + r * 32 + ((lane >> 4) << 4);
    v16bf pf = *(const v16bf*)pp;
    // --- P (16x32) * V (32x128): 8 WMMAs over d-subtiles ---
#pragma unroll
    for (int d = 0; d < 8; ++d) {
      v16bf vf = *(const v16bf*)(Vt + (size_t)(d * 16 + r) * SEQ + sb +
                                 ((lane >> 4) << 4));
      acc[d] = wmma_bf16(pf, vf, acc[d]);
    }
  }

  // epilogue: normalize and write bf16, head-concatenated token-major
  size_t token0 = (size_t)b * SEQ + qbase;
#pragma unroll
  for (int d = 0; d < 8; ++d)
#pragma unroll
    for (int e = 0; e < 8; ++e)
      oh[(token0 + h8 + e) * D_MODEL + h * DH + d * 16 + r] =
          (__bf16)(acc[d][e] / l[e]);
}

// ---------------------------------------------------------------------------
// Host launcher. Workspace layout (124 MiB total):
//   [0,16)   xb     bf16 [4096,2048]
//   [16,28)  wqkvT  bf16 [3072,2048]
//   [28,36)  woT    bf16 [2048,2048]
//   [36,84)  qkv    f32  [4096,3072]
//   [84,100) qh     bf16 [2,16,2048,128]   (pre-scaled by 1/sqrt(DH))
//   [100,104)kh     bf16 [2,4,2048,128]
//   [104,108)vT     bf16 [2,4,128,2048]    (transposed)
//   [108,124)oh     bf16 [4096,2048]
// ---------------------------------------------------------------------------
extern "C" void kernel_launch(void* const* d_in, const int* in_sizes, int n_in,
                              void* d_out, int out_size, void* d_ws,
                              size_t ws_size, hipStream_t stream) {
  (void)in_sizes; (void)n_in; (void)out_size; (void)ws_size;
  const float* x  = (const float*)d_in[0];
  const float* wq = (const float*)d_in[1];
  const float* wk = (const float*)d_in[2];
  const float* wv = (const float*)d_in[3];
  const float* wo = (const float*)d_in[4];
  const float* qw = (const float*)d_in[5];
  const float* kw = (const float*)d_in[6];

  char* ws = (char*)d_ws;
  const size_t MB = (size_t)1 << 20;
  __bf16* xb    = (__bf16*)(ws + 0 * MB);
  __bf16* wqkvT = (__bf16*)(ws + 16 * MB);
  __bf16* woT   = (__bf16*)(ws + 28 * MB);
  float*  qkv   = (float*) (ws + 36 * MB);
  __bf16* qh    = (__bf16*)(ws + 84 * MB);
  __bf16* kh    = (__bf16*)(ws + 100 * MB);
  __bf16* vT    = (__bf16*)(ws + 104 * MB);
  __bf16* oh    = (__bf16*)(ws + 108 * MB);

  cvt_bf16_kernel<<<(TOKENS * D_MODEL) / 256, 256, 0, stream>>>(
      x, xb, TOKENS * D_MODEL);
  pack_wqkvT_kernel<<<(QKV_N * D_MODEL) / 256, 256, 0, stream>>>(
      wq, wk, wv, wqkvT);
  pack_woT_kernel<<<(D_MODEL * D_MODEL) / 256, 256, 0, stream>>>(wo, woT);

  gemm_bf16_kernel<<<dim3(QKV_N / 128, TOKENS / 128), 256, 0, stream>>>(
      xb, wqkvT, qkv, QKV_N, D_MODEL);

  qkv_post_kernel<<<(TOKENS * 24) / 8, 256, 0, stream>>>(
      qkv, qw, kw, qh, kh, vT);

  attn_kernel<<<dim3(SEQ / 128, N_HEADS, BATCH), 256, 0, stream>>>(
      qh, kh, vT, oh);

  gemm_bf16_kernel<<<dim3(D_MODEL / 128, TOKENS / 128), 256, 0, stream>>>(
      oh, woT, (float*)d_out, D_MODEL, D_MODEL);
}
